// MessageDecoder_3942779978119
// MI455X (gfx1250) — compile-verified, matrix operand order checked by reading, not run on hardware
//
#include <hip/hip_runtime.h>
#include <hip/hip_bf16.h>
#include <stdint.h>
#include <stddef.h>

#define BATCH   4096
#define EDIM    1024
#define HDIM    1024
#define KCAT    2048
#define NGATES  4096
#define MSG_LEN 20
#define ALPHA   64
#define BOS_SYM 64
#define PAD_SYM 65

typedef __bf16 bf16_t;
typedef __attribute__((ext_vector_type(8)))  __bf16 v8bf;
typedef __attribute__((ext_vector_type(16))) __bf16 v16bf;
typedef __attribute__((ext_vector_type(8)))  float  v8f;

// ---------------------------------------------------------------------------
// WMMA fragment helpers (CDNA5 16x16x32 bf16 layout, wave32).
// A (16x32): lane L holds row M=L&15; element e maps to K = 8*(L>>4) +
//            (e<8 ? e : 16+(e-8))  -> two contiguous 16B chunks per lane.
// B (32x16): W stored row-major [N][K]; lane L holds column N=L&15 with the
//            same per-lane K mapping -> also two contiguous 16B chunks.
// C/D: lane L, vgpr v -> M = v + 8*(L>>4), N = L&15.
// ---------------------------------------------------------------------------
static __device__ __forceinline__ v16bf load_frag(const bf16_t* p) {
  v8bf lo = *reinterpret_cast<const v8bf*>(p);
  v8bf hi = *reinterpret_cast<const v8bf*>(p + 16);
  return __builtin_shufflevector(lo, hi, 0,1,2,3,4,5,6,7,8,9,10,11,12,13,14,15);
}

#define WMMA_BF16(A_, B_, C_) \
  __builtin_amdgcn_wmma_f32_16x16x32_bf16(false, (A_), false, (B_), (short)0, (C_), false, false)

static __device__ __forceinline__ float sigmf(float x) {
  return 1.0f / (1.0f + __expf(-x));
}

static __device__ __forceinline__ unsigned wmix(unsigned x) {
  x ^= x >> 16; x *= 0x7feb352du;
  x ^= x >> 15; x *= 0x846ca68bu;
  x ^= x >> 16; return x;
}

// ---------------------------------------------------------------------------
// Prep kernels
// ---------------------------------------------------------------------------
__global__ void k_cvt(const float* __restrict__ src, bf16_t* __restrict__ dst, int n) {
  int i = blockIdx.x * blockDim.x + threadIdx.x;
  if (i < n) dst[i] = (bf16_t)src[i];
}

// Wcat[n][k] : k<1024 -> W_ih[n][k], else W_hh[n][k-1024]
__global__ void k_cvt_cat(const float* __restrict__ Wih, const float* __restrict__ Whh,
                          bf16_t* __restrict__ Wcat) {
  int i = blockIdx.x * blockDim.x + threadIdx.x;
  if (i >= NGATES * KCAT) return;
  int n = i / KCAT, k = i % KCAT;
  float v = (k < EDIM) ? Wih[(size_t)n * EDIM + k] : Whh[(size_t)n * HDIM + (k - EDIM)];
  Wcat[i] = (bf16_t)v;
}

__global__ void k_bias_sum(const float* __restrict__ bih, const float* __restrict__ bhh,
                           float* __restrict__ bias) {
  int i = blockIdx.x * blockDim.x + threadIdx.x;
  if (i < NGATES) bias[i] = bih[i] + bhh[i];
}

__global__ void k_init_state(int* __restrict__ last, int* __restrict__ stopped) {
  int i = blockIdx.x * blockDim.x + threadIdx.x;
  if (i < BATCH) { last[i] = BOS_SYM; stopped[i] = 0; }
}

// A[:, 0:1024] = embed_bf[BOS]
__global__ void k_init_x(const bf16_t* __restrict__ embed, bf16_t* __restrict__ A) {
  int i = blockIdx.x * blockDim.x + threadIdx.x;
  if (i >= BATCH * EDIM) return;
  int b = i / EDIM, j = i % EDIM;
  A[(size_t)b * KCAT + j] = embed[(size_t)BOS_SYM * EDIM + j];
}

// ---------------------------------------------------------------------------
// Generic WMMA GEMM:  C[M][N] = A_bf16[M][K] @ W_bf16[N][K]^T + bias[N]
// OUT_BF16=0 -> fp32 out (Cf, ldc);  OUT_BF16=1 -> bf16 out (Cb, ldc)
// block = 128 threads (4 waves), each wave owns one 16x16 C tile.
// ---------------------------------------------------------------------------
template <int OUT_BF16>
__global__ void k_gemm(const bf16_t* __restrict__ A, int lda,
                       const bf16_t* __restrict__ W, int ldw,
                       const float*  __restrict__ bias,
                       float* __restrict__ Cf, bf16_t* __restrict__ Cb,
                       int ldc, int K) {
  const int wave = threadIdx.x >> 5;
  const int lane = threadIdx.x & 31;
  const int half = lane >> 4;
  const int idx  = lane & 15;
  const int ko   = half * 8;
  const int tile_n = (blockIdx.x * (blockDim.x >> 5) + wave) * 16;
  const int tile_m = blockIdx.y * 16;

  const bf16_t* pA = A + (size_t)(tile_m + idx) * lda + ko;
  const bf16_t* pB = W + (size_t)(tile_n + idx) * ldw + ko;

  v8f acc = {};
  for (int k0 = 0; k0 < K; k0 += 32) {
    v16bf a = load_frag(pA + k0);
    v16bf b = load_frag(pB + k0);
    acc = WMMA_BF16(a, b, acc);
  }

  const int n = tile_n + idx;
  const float bv = bias ? bias[n] : 0.0f;
#pragma unroll
  for (int v = 0; v < 8; ++v) {
    int m = tile_m + v + 8 * half;
    float val = acc[v] + bv;
    if (OUT_BF16) Cb[(size_t)m * ldc + n] = (bf16_t)val;
    else          Cf[(size_t)m * ldc + n] = val;
  }
}

// ---------------------------------------------------------------------------
// Fused LSTM step, M-unrolled x2: each wave computes the SAME 16-col slice of
// all 4 gate blocks (i,f,g,o) for TWO 16-row batch tiles, so the four weight
// fragments are reused twice (1.5 fragment loads per WMMA instead of 2.5,
// halving L2 weight bandwidth). The LSTM pointwise update happens in-register;
// h is written back as bf16 straight into A[:,1024:2048] (no gates buffer).
// ---------------------------------------------------------------------------
__global__ void k_lstm_step(const bf16_t* __restrict__ A,
                            const bf16_t* __restrict__ Wcat,   // [4096][2048]
                            const float*  __restrict__ bias,   // [4096]
                            float* __restrict__ c,             // [B][1024]
                            bf16_t* __restrict__ Ah) {         // == A
  const int wave = threadIdx.x >> 5;
  const int lane = threadIdx.x & 31;
  const int half = lane >> 4;
  const int idx  = lane & 15;
  const int ko   = half * 8;
  const int tile_n = (blockIdx.x * (blockDim.x >> 5) + wave) * 16;  // hidden col
  const int tile_m = blockIdx.y * 32;                               // 2 batch tiles

  const bf16_t* pA0 = A + (size_t)(tile_m + idx) * KCAT + ko;
  const bf16_t* pA1 = pA0 + (size_t)16 * KCAT;
  const bf16_t* pI  = Wcat + (size_t)(tile_n + idx) * KCAT + ko;
  const bf16_t* pF  = pI + (size_t)1024 * KCAT;
  const bf16_t* pG  = pI + (size_t)2048 * KCAT;
  const bf16_t* pO  = pI + (size_t)3072 * KCAT;

  v8f aI0 = {}, aF0 = {}, aG0 = {}, aO0 = {};
  v8f aI1 = {}, aF1 = {}, aG1 = {}, aO1 = {};
  for (int k0 = 0; k0 < KCAT; k0 += 32) {
    v16bf a0 = load_frag(pA0 + k0);
    v16bf a1 = load_frag(pA1 + k0);
    v16bf bi = load_frag(pI + k0);
    v16bf bf = load_frag(pF + k0);
    v16bf bg = load_frag(pG + k0);
    v16bf bo = load_frag(pO + k0);
    if (k0 + 64 < KCAT) {   // keep weight streams warm (global_prefetch_b8)
      __builtin_prefetch(pI + k0 + 64);
      __builtin_prefetch(pF + k0 + 64);
      __builtin_prefetch(pG + k0 + 64);
      __builtin_prefetch(pO + k0 + 64);
    }
    aI0 = WMMA_BF16(a0, bi, aI0);  aI1 = WMMA_BF16(a1, bi, aI1);
    aF0 = WMMA_BF16(a0, bf, aF0);  aF1 = WMMA_BF16(a1, bf, aF1);
    aG0 = WMMA_BF16(a0, bg, aG0);  aG1 = WMMA_BF16(a1, bg, aG1);
    aO0 = WMMA_BF16(a0, bo, aO0);  aO1 = WMMA_BF16(a1, bo, aO1);
  }

  const int j  = tile_n + idx;
  const float bi_ = bias[j];
  const float bf_ = bias[1024 + j];
  const float bg_ = bias[2048 + j];
  const float bo_ = bias[3072 + j];
#pragma unroll
  for (int v = 0; v < 8; ++v) {
    // tile 0
    {
      int m = tile_m + v + 8 * half;
      float iv = sigmf(aI0[v] + bi_);
      float fv = sigmf(aF0[v] + bf_);
      float gv = tanhf(aG0[v] + bg_);
      float ov = sigmf(aO0[v] + bo_);
      size_t ci = (size_t)m * HDIM + j;
      float c2 = fv * c[ci] + iv * gv;
      float h2 = ov * tanhf(c2);
      c[ci] = c2;
      Ah[(size_t)m * KCAT + EDIM + j] = (bf16_t)h2;
    }
    // tile 1
    {
      int m = tile_m + 16 + v + 8 * half;
      float iv = sigmf(aI1[v] + bi_);
      float fv = sigmf(aF1[v] + bf_);
      float gv = tanhf(aG1[v] + bg_);
      float ov = sigmf(aO1[v] + bo_);
      size_t ci = (size_t)m * HDIM + j;
      float c2 = fv * c[ci] + iv * gv;
      float h2 = ov * tanhf(c2);
      c[ci] = c2;
      Ah[(size_t)m * KCAT + EDIM + j] = (bf16_t)h2;
    }
  }
}

// ---------------------------------------------------------------------------
// Per-row softmax stats + Gumbel-max categorical sample + next-x gather.
// One block (64 threads, 2 waves) per batch row.
// ---------------------------------------------------------------------------
__global__ void k_sample(const float* __restrict__ logits,   // [B][64]
                         const bf16_t* __restrict__ embed,   // [66][1024]
                         int* __restrict__ last, int* __restrict__ stopped,
                         float* __restrict__ out_lp,         // [B][20]
                         float* __restrict__ out_msg,        // [B][20]
                         float* __restrict__ ents,           // [B][20]
                         bf16_t* __restrict__ Ax,            // A (x half)
                         int t) {
  __shared__ float sl[ALPHA];
  __shared__ float red[ALPHA];
  __shared__ int   ridx[ALPHA];
  __shared__ int   s_action;
  __shared__ float s_logZ;

  const int b = blockIdx.x;
  const int tid = threadIdx.x;
  float l = logits[(size_t)b * ALPHA + tid];
  sl[tid] = l;

  // max
  red[tid] = l;
  for (int s = 32; s > 0; s >>= 1) {
    __syncthreads();
    if (tid < s) red[tid] = fmaxf(red[tid], red[tid + s]);
  }
  __syncthreads();
  float mx = red[0];
  __syncthreads();

  // sum exp
  red[tid] = __expf(l - mx);
  for (int s = 32; s > 0; s >>= 1) {
    __syncthreads();
    if (tid < s) red[tid] += red[tid + s];
  }
  __syncthreads();
  float logZ = __logf(red[0]) + mx;
  if (tid == 0) s_logZ = logZ;
  __syncthreads();

  // sum p*logp
  float logp = l - logZ;
  red[tid] = __expf(logp) * logp;
  for (int s = 32; s > 0; s >>= 1) {
    __syncthreads();
    if (tid < s) red[tid] += red[tid + s];
  }
  __syncthreads();
  float plogp_sum = red[0];
  __syncthreads();

  // Gumbel-max argmax
  unsigned r = wmix(wmix(wmix(0x2545F491u ^ (unsigned)t * 0x9E3779B9u)
                         ^ (unsigned)b * 0x85EBCA77u)
                    ^ (unsigned)tid * 0xC2B2AE3Du);
  float u = (float)(r >> 8) * 0x1p-24f + 0x1p-25f;
  float key = l + (-__logf(-__logf(u)));
  red[tid] = key; ridx[tid] = tid;
  for (int s = 32; s > 0; s >>= 1) {
    __syncthreads();
    if (tid < s && red[tid + s] > red[tid]) { red[tid] = red[tid + s]; ridx[tid] = ridx[tid + s]; }
  }
  __syncthreads();

  if (tid == 0) {
    int act = ridx[0];
    int stp = stopped[b];
    float live = stp ? 0.0f : 1.0f;
    out_lp[(size_t)b * MSG_LEN + t]  = (sl[act] - s_logZ) * live;
    ents[(size_t)b * MSG_LEN + t]   = -plogp_sum * live;
    int act2 = stp ? PAD_SYM : act;
    stopped[b] = stp | (act2 == 0);    // EOS == 0
    last[b] = act2;
    out_msg[(size_t)b * MSG_LEN + t] = (float)act2;
    s_action = act2;
  }
  __syncthreads();

  // next-step embedding: A[b, 0:1024] = embed[action]  (embed[PAD] row is zero)
  const int a = s_action;
  for (int j = tid; j < EDIM; j += ALPHA)
    Ax[(size_t)b * KCAT + j] = embed[(size_t)a * EDIM + j];
}

__global__ void k_finalize(const float* __restrict__ msg,   // [B][20]
                           const float* __restrict__ ents,  // [B][20]
                           float* __restrict__ out_ent,     // [B]
                           float* __restrict__ out_len) {   // [B]
  int b = blockIdx.x * blockDim.x + threadIdx.x;
  if (b >= BATCH) return;
  int cnt = 0; float s = 0.0f;
  for (int t = 0; t < MSG_LEN; ++t) {
    cnt += (msg[(size_t)b * MSG_LEN + t] != (float)PAD_SYM) ? 1 : 0;
    s   += ents[(size_t)b * MSG_LEN + t];
  }
  out_ent[b] = s / (float)cnt;
  out_len[b] = (float)cnt;
}

// ---------------------------------------------------------------------------
static inline size_t alignup(size_t x) { return (x + 255) & ~(size_t)255; }

extern "C" void kernel_launch(void* const* d_in, const int* in_sizes, int n_in,
                              void* d_out, int out_size, void* d_ws, size_t ws_size,
                              hipStream_t stream) {
  (void)in_sizes; (void)n_in; (void)out_size; (void)ws_size;
  const float* encoded = (const float*)d_in[0];
  const float* embed   = (const float*)d_in[1];
  const float* W_ih    = (const float*)d_in[2];
  const float* W_hh    = (const float*)d_in[3];
  const float* b_ih    = (const float*)d_in[4];
  const float* b_hh    = (const float*)d_in[5];
  const float* W_cell  = (const float*)d_in[6];
  const float* b_cell  = (const float*)d_in[7];
  const float* W_hid   = (const float*)d_in[8];
  const float* b_hid   = (const float*)d_in[9];
  const float* W_act   = (const float*)d_in[10];
  const float* b_act   = (const float*)d_in[11];

  // workspace carve-up
  char* p = (char*)d_ws;
  size_t off = 0;
  auto take = [&](size_t bytes) { char* q = p + off; off = alignup(off + bytes); return q; };
  bf16_t* Wcat   = (bf16_t*)take((size_t)NGATES * KCAT * 2);
  bf16_t* Wcellb = (bf16_t*)take((size_t)EDIM * EDIM * 2);
  bf16_t* Whidb  = (bf16_t*)take((size_t)EDIM * EDIM * 2);
  bf16_t* Wactb  = (bf16_t*)take((size_t)ALPHA * EDIM * 2);
  bf16_t* embedb = (bf16_t*)take((size_t)(ALPHA + 2) * EDIM * 2);
  bf16_t* encb   = (bf16_t*)take((size_t)BATCH * EDIM * 2);
  float*  bias   = (float*) take((size_t)NGATES * 4);
  bf16_t* Abuf   = (bf16_t*)take((size_t)BATCH * KCAT * 2);   // [x|h] bf16
  float*  cbuf   = (float*) take((size_t)BATCH * HDIM * 4);
  float*  logits = (float*) take((size_t)BATCH * ALPHA * 4);
  int*    last   = (int*)   take((size_t)BATCH * 4);
  int*    stop_  = (int*)   take((size_t)BATCH * 4);
  float*  ents   = (float*) take((size_t)BATCH * MSG_LEN * 4);

  float* out      = (float*)d_out;
  float* out_ent  = out;                                   // [B,1]
  float* out_lp   = out + BATCH;                           // [B,20]
  float* out_msg  = out + BATCH + BATCH * MSG_LEN;         // [B,20]
  float* out_len  = out + BATCH + 2 * BATCH * MSG_LEN;     // [B,1]

  // ---- prep ----
  {
    int n = NGATES * KCAT;
    k_cvt_cat<<<(n + 255) / 256, 256, 0, stream>>>(W_ih, W_hh, Wcat);
  }
  k_cvt<<<(EDIM * EDIM + 255) / 256, 256, 0, stream>>>(W_cell, Wcellb, EDIM * EDIM);
  k_cvt<<<(EDIM * EDIM + 255) / 256, 256, 0, stream>>>(W_hid,  Whidb,  EDIM * EDIM);
  k_cvt<<<(ALPHA * EDIM + 255) / 256, 256, 0, stream>>>(W_act, Wactb, ALPHA * EDIM);
  k_cvt<<<((ALPHA + 2) * EDIM + 255) / 256, 256, 0, stream>>>(embed, embedb, (ALPHA + 2) * EDIM);
  k_cvt<<<(BATCH * EDIM + 255) / 256, 256, 0, stream>>>(encoded, encb, BATCH * EDIM);
  k_bias_sum<<<(NGATES + 255) / 256, 256, 0, stream>>>(b_ih, b_hh, bias);
  k_init_state<<<(BATCH + 255) / 256, 256, 0, stream>>>(last, stop_);
  k_init_x<<<(BATCH * EDIM + 255) / 256, 256, 0, stream>>>(embedb, Abuf);

  // ---- h0 = enc @ W_cell^T + b_cell (bf16, into A[:,1024:]) ----
  k_gemm<1><<<dim3(EDIM / 64, BATCH / 16), 128, 0, stream>>>(
      encb, EDIM, Wcellb, EDIM, b_cell, nullptr, Abuf + EDIM, KCAT, EDIM);
  // ---- c0 = enc @ W_hid^T + b_hid (fp32) ----
  k_gemm<0><<<dim3(EDIM / 64, BATCH / 16), 128, 0, stream>>>(
      encb, EDIM, Whidb, EDIM, b_hid, cbuf, nullptr, HDIM, EDIM);

  // ---- 20 decode steps ----
  for (int t = 0; t < MSG_LEN; ++t) {
    k_lstm_step<<<dim3(HDIM / 64, BATCH / 32), 128, 0, stream>>>(
        Abuf, Wcat, bias, cbuf, Abuf);
    k_gemm<0><<<dim3(ALPHA / 64, BATCH / 16), 128, 0, stream>>>(
        Abuf + EDIM, KCAT, Wactb, EDIM, b_act, logits, nullptr, ALPHA, EDIM);
    k_sample<<<BATCH, ALPHA, 0, stream>>>(
        logits, embedb, last, stop_, out_lp, out_msg, ents, Abuf, t);
  }

  k_finalize<<<(BATCH + 255) / 256, 256, 0, stream>>>(out_msg, ents, out_ent, out_len);
}
